// FlowEmbedding_88201448391141
// MI455X (gfx1250) — matrix operand order repaired and checked by reading, not compiled, not run on hardware
//
#include <hip/hip_runtime.h>

typedef __attribute__((ext_vector_type(16))) _Float16 v16h;
typedef __attribute__((ext_vector_type(8)))  _Float16 v8h;
typedef __attribute__((ext_vector_type(8)))  float    v8f;

union V16H { v16h v; v8h h[2]; };

#define BATCH 8
#define NPTS  4096
#define CH    128
#define KNN   16
#define PTOT  65536              // NPTS * KNN per batch
#define TILE_P 128
#define BLK_PER_B 512            // PTOT / TILE_P
#define NEG_SLOPE 0.1f

// ---------------------------------------------------------------------------
// [B][C][N] f32  ->  [B][N][C] f16  (tiled, coalesced both sides)
// ---------------------------------------------------------------------------
__global__ __launch_bounds__(256) void transpose_f32_to_f16(
    const float* __restrict__ in, _Float16* __restrict__ out) {
  __shared__ float tile[32][33];
  int b = blockIdx.z;
  int n0 = blockIdx.x * 32, c0 = blockIdx.y * 32;
  const float* src = in + ((size_t)b * CH + c0) * NPTS + n0;
  for (int i = threadIdx.y; i < 32; i += 8)
    tile[i][threadIdx.x] = src[(size_t)i * NPTS + threadIdx.x];
  __syncthreads();
  _Float16* dst = out + ((size_t)b * NPTS + n0) * CH + c0;
  for (int i = threadIdx.y; i < 32; i += 8)
    dst[(size_t)i * CH + threadIdx.x] = (_Float16)tile[threadIdx.x][i];
}

// ---------------------------------------------------------------------------
// kNN: one wave per query point. xyz2 staged in LDS. Per-lane sorted top-16
// over 128 candidates, then wave-wide 16-round 64-bit min merge.
// ---------------------------------------------------------------------------
__global__ __launch_bounds__(256) void knn_kernel(
    const float* __restrict__ xyz1, const float* __restrict__ xyz2,
    int* __restrict__ kidx) {
  __shared__ float p2[3 * NPTS];                       // 48 KB
  int b  = blockIdx.x >> 9;                            // 512 blocks per batch
  int n0 = (blockIdx.x & 511) * 8;
  const float* x2 = xyz2 + (size_t)b * 3 * NPTS;
  for (int j = threadIdx.x; j < 3 * NPTS; j += 256) p2[j] = x2[j];
  __syncthreads();

  int wave = threadIdx.x >> 5, lane = threadIdx.x & 31;
  int n = n0 + wave;
  const float* x1 = xyz1 + (size_t)b * 3 * NPTS + n;
  float qx = x1[0], qy = x1[NPTS], qz = x1[2 * NPTS];

  float d[16]; int id[16];
#pragma unroll
  for (int i = 0; i < 16; ++i) { d[i] = 3.4e38f; id[i] = 0; }

  for (int m0 = 0; m0 < NPTS; m0 += 32) {
    int m = m0 + lane;
    float dx = p2[m] - qx, dy = p2[NPTS + m] - qy, dz = p2[2 * NPTS + m] - qz;
    float dd = dx * dx + dy * dy + dz * dz;
    if (dd < d[15]) {                                  // sorted insert (registers)
      d[15] = dd; id[15] = m;
#pragma unroll
      for (int j = 15; j > 0; --j) {
        if (d[j] < d[j - 1]) {
          float td = d[j]; d[j] = d[j - 1]; d[j - 1] = td;
          int   ti = id[j]; id[j] = id[j - 1]; id[j - 1] = ti;
        }
      }
    }
  }
  // 32-way merge of per-lane sorted lists: pop-with-shift keeps indices static
  int* dst = kidx + ((size_t)(b * NPTS + n)) * KNN;
  for (int i = 0; i < KNN; ++i) {
    unsigned fu = __float_as_uint(d[0]);               // d >= 0 -> monotonic
    unsigned long long pk =
        ((unsigned long long)fu << 32) | (unsigned)((lane << 12) | id[0]);
#pragma unroll
    for (int o = 16; o > 0; o >>= 1) {
      unsigned long long q = __shfl_xor(pk, o, 32);
      pk = q < pk ? q : pk;
    }
    int wl = (int)((pk >> 12) & 31u);
    if (lane == wl) {                                  // winner pops its head
#pragma unroll
      for (int j = 0; j < 15; ++j) { d[j] = d[j + 1]; id[j] = id[j + 1]; }
      d[15] = 3.4e38f; id[15] = 0;
    }
    if (lane == 0) dst[i] = (int)(pk & 0xFFFu);
  }
}

// ---------------------------------------------------------------------------
// WMMA GEMM tile: 128 points x 128 out-channels per block, 8 waves.
// Wave w -> co rows 16w..16w+15 == GroupNorm group w.
// A: 16x32 f16 layout (lane=M; K 0-7/8-15 split by lane half, +16 in hi VGPRs)
// B: from LDS [point][channel], two 16B ds loads per fragment.
// D/C: VGPR r -> (M = r + 8*hi, N = lane&15).
// ---------------------------------------------------------------------------
template <int S, int NK, int CI>
__device__ __forceinline__ void gemm_tile(
    const _Float16* xs, const float* __restrict__ W,
    const float* __restrict__ bias, _Float16* __restrict__ ybuf,
    float* __restrict__ partial, int b, int P0, int wave, int lane, int blockId) {
  int m  = lane & 15;
  int hi = lane >> 4;
  int co0 = wave * 16;
  int kbA = hi * 8;

  V16H a[NK];
#pragma unroll
  for (int kc = 0; kc < NK; ++kc) {
    int k0 = kc * 32;
#pragma unroll
    for (int e = 0; e < 8; ++e) {
      int kA = k0 + kbA + e;
      int kB = k0 + 16 + kbA + e;
      float va = (kA < CI) ? W[(size_t)(co0 + m) * CI + kA] : 0.f;
      float vb = (kB < CI) ? W[(size_t)(co0 + m) * CI + kB] : 0.f;
      a[kc].v[e]     = (_Float16)va;
      a[kc].v[8 + e] = (_Float16)vb;
    }
  }
  float bv[8];
#pragma unroll
  for (int r = 0; r < 8; ++r) bv[r] = bias[co0 + r + hi * 8];

  float s = 0.f, s2 = 0.f;
  int kbB = hi * 8;
  for (int pt = 0; pt < 8; ++pt) {
    v8f acc = {};
#pragma unroll
    for (int kc = 0; kc < NK; ++kc) {
      const _Float16* src = xs + (pt * 16 + m) * S + kc * 32 + kbB;
      V16H bf;
      bf.h[0] = *(const v8h*)(src);
      bf.h[1] = *(const v8h*)(src + 16);
      acc = __builtin_amdgcn_wmma_f32_16x16x32_f16(
          false, a[kc].v, false, bf.v, (short)0, acc, false, false);
    }
    size_t pcol = (size_t)P0 + pt * 16 + m;
#pragma unroll
    for (int r = 0; r < 8; ++r) {
      float y = acc[r] + bv[r];
      s += y; s2 += y * y;
      ybuf[((size_t)(b * CH + co0 + r + hi * 8)) * PTOT + pcol] = (_Float16)y;
    }
  }
#pragma unroll
  for (int o = 16; o > 0; o >>= 1) {
    s  += __shfl_xor(s, o, 32);
    s2 += __shfl_xor(s2, o, 32);
  }
  if (lane == 0) {                                     // deterministic partials
    partial[(size_t)(blockId * 8 + wave) * 2 + 0] = s;
    partial[(size_t)(blockId * 8 + wave) * 2 + 1] = s2;
  }
}

// ---------------------------------------------------------------------------
// Layer 0: build concat(feat1, feat2[knn], dxyz, zero-pad) tile directly in
// LDS from pre-transposed f16 feature rows, then WMMA with Ci padded 259->288.
// ---------------------------------------------------------------------------
__global__ __launch_bounds__(256) void conv0_kernel(
    const _Float16* __restrict__ f1T, const _Float16* __restrict__ f2T,
    const float* __restrict__ xyz1, const float* __restrict__ xyz2,
    const int* __restrict__ kidx, const float* __restrict__ W,
    const float* __restrict__ bias, _Float16* __restrict__ ybuf,
    float* __restrict__ partial) {
  extern __shared__ _Float16 xs[];                     // [128][296]
  const int S = 296;
  int b  = blockIdx.x >> 9;
  int P0 = (blockIdx.x & 511) * TILE_P;
  int wave = threadIdx.x >> 5, lane = threadIdx.x & 31;

  for (int p = wave * 16; p < wave * 16 + 16; ++p) {
    int P = P0 + p, n = P >> 4, kk = P & 15;
    int idx = kidx[((size_t)(b * NPTS + n)) * KNN + kk];
    const _Float16* r1 = f1T + ((size_t)(b * NPTS + n)) * CH;
    const _Float16* r2 = f2T + ((size_t)(b * NPTS + idx)) * CH;
    int c4 = lane * 4;                                 // 8B per lane, coalesced
    *(uint2*)(&xs[p * S + c4])      = *(const uint2*)(r1 + c4);
    *(uint2*)(&xs[p * S + CH + c4]) = *(const uint2*)(r2 + c4);
    if (lane < 3) {
      float dv = xyz2[(size_t)b * 3 * NPTS + lane * NPTS + idx]
               - xyz1[(size_t)b * 3 * NPTS + lane * NPTS + n];
      xs[p * S + 256 + lane] = (_Float16)dv;
    } else {
      xs[p * S + 256 + lane] = (_Float16)0.f;          // 259..287 zero pad
    }
  }
  __syncthreads();
  gemm_tile<296, 9, 259>(xs, W, bias, ybuf, partial, b, P0, wave, lane, blockIdx.x);
}

// ---------------------------------------------------------------------------
// Layers 1/2: fused GroupNorm(prev) + LeakyReLU on LDS fill, then WMMA.
// In-place on ybuf: all reads precede __syncthreads(); each block writes only
// its own point columns.
// ---------------------------------------------------------------------------
__global__ __launch_bounds__(256) void convN_kernel(
    const _Float16* __restrict__ xbuf, const float* __restrict__ W,
    const float* __restrict__ bias, const float* __restrict__ mstd_prev,
    const float* __restrict__ gamma_prev, const float* __restrict__ beta_prev,
    _Float16* __restrict__ ybuf, float* __restrict__ partial) {
  extern __shared__ _Float16 xs[];                     // [128][136]
  const int S = 136;
  int b  = blockIdx.x >> 9;
  int P0 = (blockIdx.x & 511) * TILE_P;
  int c = threadIdx.x >> 1, h = threadIdx.x & 1;
  {
    float mean = mstd_prev[(b * 8 + (c >> 4)) * 2 + 0];
    float rstd = mstd_prev[(b * 8 + (c >> 4)) * 2 + 1];
    float g  = gamma_prev[c] * rstd;
    float bt = beta_prev[c] - mean * g;                // y = x*g + bt
    const _Float16* src = xbuf + ((size_t)(b * CH + c)) * PTOT + P0 + h * 64;
#pragma unroll
    for (int i = 0; i < 64; ++i) {
      float v = (float)src[i] * g + bt;
      v = v >= 0.f ? v : NEG_SLOPE * v;                // LeakyReLU
      xs[(h * 64 + i) * S + c] = (_Float16)v;
    }
  }
  __syncthreads();
  int wave = threadIdx.x >> 5, lane = threadIdx.x & 31;
  gemm_tile<136, 4, 128>(xs, W, bias, ybuf, partial, b, P0, wave, lane, blockIdx.x);
}

// ---------------------------------------------------------------------------
// Reduce per-block partials -> mean / rstd per (batch, group). Deterministic.
// ---------------------------------------------------------------------------
__global__ void finalize_kernel(const float* __restrict__ partial,
                                float* __restrict__ mstd) {
  int t = threadIdx.x;                                 // 64 = b*8 + g
  int b = t >> 3, g = t & 7;
  float s = 0.f, s2 = 0.f;
  for (int blk = 0; blk < BLK_PER_B; ++blk) {
    size_t base = ((size_t)((b * BLK_PER_B + blk) * 8 + g)) * 2;
    s += partial[base]; s2 += partial[base + 1];
  }
  const float inv = 1.f / (16.f * (float)PTOT);
  float mean = s * inv;
  float var  = s2 * inv - mean * mean;
  mstd[t * 2 + 0] = mean;
  mstd[t * 2 + 1] = rsqrtf(var + 1e-5f);
}

// ---------------------------------------------------------------------------
// Final: GN(layer2) + LeakyReLU + max over K -> [B][C][N] f32
// ---------------------------------------------------------------------------
__global__ __launch_bounds__(256) void final_kernel(
    const _Float16* __restrict__ ybuf, const float* __restrict__ mstd,
    const float* __restrict__ gamma, const float* __restrict__ beta,
    float* __restrict__ out) {
  size_t t = (size_t)blockIdx.x * 256 + threadIdx.x;   // b*2^19 + c*2^12 + n
  int n = (int)(t & 4095);
  int c = (int)((t >> 12) & 127);
  int b = (int)(t >> 19);
  float mean = mstd[(b * 8 + (c >> 4)) * 2 + 0];
  float rstd = mstd[(b * 8 + (c >> 4)) * 2 + 1];
  float g  = gamma[c] * rstd;
  float bt = beta[c] - mean * g;
  const _Float16* src = ybuf + ((size_t)(b * CH + c)) * PTOT + (size_t)n * KNN;
  float mx = -3.4e38f;
#pragma unroll
  for (int i = 0; i < KNN; ++i) {
    float v = (float)src[i] * g + bt;
    v = v >= 0.f ? v : NEG_SLOPE * v;
    mx = v > mx ? v : mx;
  }
  out[t] = mx;
}

// ---------------------------------------------------------------------------
extern "C" void kernel_launch(void* const* d_in, const int* in_sizes, int n_in,
                              void* d_out, int out_size, void* d_ws, size_t ws_size,
                              hipStream_t stream) {
  const float* xyz1  = (const float*)d_in[0];
  const float* xyz2  = (const float*)d_in[1];
  const float* feat1 = (const float*)d_in[2];
  const float* feat2 = (const float*)d_in[3];
  const float* W0 = (const float*)d_in[4];  const float* b0 = (const float*)d_in[5];
  const float* g0 = (const float*)d_in[6];  const float* bt0 = (const float*)d_in[7];
  const float* W1 = (const float*)d_in[8];  const float* b1 = (const float*)d_in[9];
  const float* g1 = (const float*)d_in[10]; const float* bt1 = (const float*)d_in[11];
  const float* W2 = (const float*)d_in[12]; const float* b2 = (const float*)d_in[13];
  const float* g2 = (const float*)d_in[14]; const float* bt2 = (const float*)d_in[15];

  char* ws = (char*)d_ws;
  size_t off = 0;
  int* kidx = (int*)(ws + off);           off += (size_t)BATCH * NPTS * KNN * 4;   // 2 MB
  _Float16* f1T = (_Float16*)(ws + off);  off += (size_t)BATCH * NPTS * CH * 2;    // 8 MB
  _Float16* f2T = (_Float16*)(ws + off);  off += (size_t)BATCH * NPTS * CH * 2;    // 8 MB
  float* partial = (float*)(ws + off);    off += (size_t)4096 * 8 * 2 * 4;         // 256 KB
  float* mstd = (float*)(ws + off);       off += 3 * 64 * 2 * 4;
  off = (off + 255) & ~(size_t)255;
  _Float16* ybuf = (_Float16*)(ws + off); off += (size_t)BATCH * CH * PTOT * 2;    // 128 MB

  dim3 tb(32, 8);
  transpose_f32_to_f16<<<dim3(NPTS / 32, CH / 32, BATCH), tb, 0, stream>>>(feat1, f1T);
  transpose_f32_to_f16<<<dim3(NPTS / 32, CH / 32, BATCH), tb, 0, stream>>>(feat2, f2T);
  knn_kernel<<<4096, 256, 0, stream>>>(xyz1, xyz2, kidx);

  conv0_kernel<<<4096, 256, 128 * 296 * 2, stream>>>(f1T, f2T, xyz1, xyz2, kidx,
                                                     W0, b0, ybuf, partial);
  finalize_kernel<<<1, 64, 0, stream>>>(partial, mstd + 0);
  convN_kernel<<<4096, 256, 128 * 136 * 2, stream>>>(ybuf, W1, b1, mstd + 0, g0, bt0,
                                                     ybuf, partial);
  finalize_kernel<<<1, 64, 0, stream>>>(partial, mstd + 128);
  convN_kernel<<<4096, 256, 128 * 136 * 2, stream>>>(ybuf, W2, b2, mstd + 128, g1, bt1,
                                                     ybuf, partial);
  finalize_kernel<<<1, 64, 0, stream>>>(partial, mstd + 256);
  final_kernel<<<(out_size + 255) / 256, 256, 0, stream>>>(ybuf, mstd + 256, g2, bt2,
                                                           (float*)d_out);
}
